// SwinTransformer_33251636805785
// MI455X (gfx1250) — matrix operand order repaired and checked by reading, not compiled, not run on hardware
//
#include <hip/hip_runtime.h>
#include <hip/hip_bf16.h>
#include <math.h>

typedef __attribute__((ext_vector_type(16))) __bf16 v16bf;
typedef __attribute__((ext_vector_type(8)))  float  v8f;

#define BB   64
#define HH   56
#define WW   56
#define CC   128
#define NHD  4
#define WSZ  7
#define NTK  49
#define NWIN 64
#define MLPD 512
#define HDD  32
#define TOK  200704            // B*H*W
#define SCALE 0.17677669529663689f

// ---------------- helpers ----------------

__device__ __forceinline__ v8f zero8() {
    v8f z;
#pragma unroll
    for (int i = 0; i < 8; ++i) z[i] = 0.0f;
    return z;
}

__device__ __forceinline__ v8f wmma_bf16(v16bf a, v16bf b, v8f c) {
    return __builtin_amdgcn_wmma_f32_16x16x32_bf16(false, a, false, b, (short)0, c, false, false);
}

union FragU { uint4 u4[2]; v16bf v; };

// windowed token index -> natural token index (shift fold: nat = win_coord + 3 mod 56)
__device__ __forceinline__ long nat_from_win(int mw) {
    int win = mw / NTK, t = mw - win * NTK;
    int b = win >> 6, wrem = win & 63;
    int wy = wrem >> 3, wx = wrem & 7;
    int ty = t / WSZ, tx = t - ty * WSZ;
    int hn = wy * WSZ + ty + 3; if (hn >= HH) hn -= HH;
    int wn = wx * WSZ + tx + 3; if (wn >= WW) wn -= WW;
    return (long)b * (HH * WW) + hn * WW + wn;
}

// ---------------- weight pack: w (Nout,K) fp32 -> B fragments bf16 ----------------
// layout: [N/16][K/32][lane 0..31][16 elems]; lane holds col n=lane%16,
// K = kt*32 + (lane/16)*16 + e   (e = 0..15)
__global__ void pack_b_kernel(const float* __restrict__ w, __bf16* __restrict__ out,
                              int Nout, int K) {
    int idx = blockIdx.x * blockDim.x + threadIdx.x;
    if (idx >= Nout * K) return;
    int perNT = 16 * K;
    int nt   = idx / perNT;
    int rem  = idx - nt * perNT;
    int kt   = rem >> 9;            // rem / 512
    int rem2 = rem & 511;
    int lane = rem2 >> 4;
    int e    = rem2 & 15;
    int k = kt * 32 + ((lane >> 4) << 4) + e;
    int n = nt * 16 + (lane & 15);
    out[idx] = (__bf16)w[(long)n * K + k];
}

// ---------------- fused (mask + rel-pos bias) table: cb[widx][h][m][n] ----------------
__global__ void fuse_bias_kernel(const float* __restrict__ mask,
                                 const int* __restrict__ rel_idx,
                                 const float* __restrict__ rel_table,
                                 float* __restrict__ cb) {
    int idx = blockIdx.x * blockDim.x + threadIdx.x;   // 64*4*49*49 = 614656
    if (idx >= NWIN * NHD * NTK * NTK) return;
    int mn   = idx % (NTK * NTK);
    int rest = idx / (NTK * NTK);
    int h    = rest & 3;
    int widx = rest >> 2;
    cb[idx] = mask[widx * (NTK * NTK) + mn] + rel_table[rel_idx[mn] * NHD + h];
}

// ---------------- layernorm (wave per token), optional window permute ----------------
__global__ void ln_kernel(const float* __restrict__ x, const float* __restrict__ g,
                          const float* __restrict__ bta, __bf16* __restrict__ out,
                          int permute) {
    int lane = threadIdx.x & 31;
    int wv   = threadIdx.x >> 5;
    int tid  = blockIdx.x * 8 + wv;           // windowed (or flat) token id
    if (tid >= TOK) return;
    long src = permute ? nat_from_win(tid) : (long)tid;

    float4 v = ((const float4*)(x + src * CC))[lane];
    float s = v.x + v.y + v.z + v.w;
#pragma unroll
    for (int m = 16; m >= 1; m >>= 1) s += __shfl_xor(s, m, 32);
    float mu = s * (1.0f / CC);
    float dx = v.x - mu, dy = v.y - mu, dz = v.z - mu, dw = v.w - mu;
    float q = dx * dx + dy * dy + dz * dz + dw * dw;
#pragma unroll
    for (int m = 16; m >= 1; m >>= 1) q += __shfl_xor(q, m, 32);
    float rs = rsqrtf(q * (1.0f / CC) + 1e-5f);

    float4 gg = ((const float4*)g)[lane];
    float4 bb = ((const float4*)bta)[lane];
    union { ushort4 u; __bf16 h[4]; } o;
    o.h[0] = (__bf16)(dx * rs * gg.x + bb.x);
    o.h[1] = (__bf16)(dy * rs * gg.y + bb.y);
    o.h[2] = (__bf16)(dz * rs * gg.z + bb.z);
    o.h[3] = (__bf16)(dw * rs * gg.w + bb.w);
    ((ushort4*)out)[(long)tid * 32 + lane] = o.u;
}

// ---------------- WMMA GEMM: C = A(bf16, MxK row-major) * Bpacked + bias -------------
// wave computes 2 M-tiles x 4 N-tiles (32x64); block = 8 waves = 128x128 tile.
// EPI: 0=qkv route  1=proj residual-permute  2=gelu->bf16  3=inplace residual fp32
template <int EPI>
__global__ __launch_bounds__(256) void
gemm_bf16(const __bf16* __restrict__ A, const __bf16* __restrict__ Bp,
          const float* __restrict__ bias, int N, int K,
          void* e0, void* e1, void* e2) {
    int lane = threadIdx.x & 31;
    int wv   = threadIdx.x >> 5;
    int half = lane >> 4, l16 = lane & 15;
    int mw = wv >> 1, nw = wv & 1;
    int mt0 = blockIdx.x * 8 + mw * 2;      // base M tile (16-row units)
    int nt0 = blockIdx.y * 8 + nw * 4;      // base N tile (16-col units)
    const int KT = K >> 5;

    v8f acc[2][4];
#pragma unroll
    for (int mi = 0; mi < 2; ++mi)
#pragma unroll
        for (int nj = 0; nj < 4; ++nj) acc[mi][nj] = zero8();

    const __bf16* arow0 = A + (long)(mt0 * 16 + l16) * K;
    const __bf16* arow1 = arow0 + (long)16 * K;

#pragma unroll 4
    for (int kt = 0; kt < KT; ++kt) {
        int kb = kt << 5;
        FragU a0, a1;
        a0.u4[0] = *(const uint4*)(arow0 + kb + half * 16);
        a0.u4[1] = *(const uint4*)(arow0 + kb + half * 16 + 8);
        a1.u4[0] = *(const uint4*)(arow1 + kb + half * 16);
        a1.u4[1] = *(const uint4*)(arow1 + kb + half * 16 + 8);
        FragU b[4];
#pragma unroll
        for (int nj = 0; nj < 4; ++nj) {
            const __bf16* bp = Bp + (((long)(nt0 + nj) * KT + kt) * 32 + lane) * 16;
            b[nj].u4[0] = ((const uint4*)bp)[0];
            b[nj].u4[1] = ((const uint4*)bp)[1];
        }
#pragma unroll
        for (int nj = 0; nj < 4; ++nj) {
            acc[0][nj] = wmma_bf16(a0.v, b[nj].v, acc[0][nj]);
            acc[1][nj] = wmma_bf16(a1.v, b[nj].v, acc[1][nj]);
        }
    }

    // ---- epilogue: per-column constants hoisted out of the row loops ----
    int   n4[4];
    float bv4[4];
#pragma unroll
    for (int nj = 0; nj < 4; ++nj) {
        n4[nj]  = (nt0 + nj) * 16 + l16;
        bv4[nj] = bias[n4[nj]];
    }

    if constexpr (EPI == 0) {               // qkv: route to q/k/v padded buffers
        __bf16* dstp[4];
        long    hoff[4];
        float   scl[4];
#pragma unroll
        for (int nj = 0; nj < 4; ++nj) {
            int which = n4[nj] >> 7, h = (n4[nj] >> 5) & 3, hd = n4[nj] & 31;
            dstp[nj] = (which == 0) ? (__bf16*)e0 : (which == 1) ? (__bf16*)e1 : (__bf16*)e2;
            hoff[nj] = (long)h * (64 * HDD) + hd;
            scl[nj]  = (which == 0) ? SCALE : 1.0f;
        }
#pragma unroll
        for (int mi = 0; mi < 2; ++mi)
#pragma unroll
            for (int r = 0; r < 8; ++r) {
                int m = (mt0 + mi) * 16 + half * 8 + r;
                int win = m / NTK, t = m - win * NTK;
                long base0 = ((long)win * (NHD * 64) + t) * HDD;
#pragma unroll
                for (int nj = 0; nj < 4; ++nj)
                    dstp[nj][base0 + hoff[nj]] =
                        (__bf16)((acc[mi][nj][r] + bv4[nj]) * scl[nj]);
            }
    } else if constexpr (EPI == 1) {        // proj: window-reverse + residual
        const float* xin = (const float*)e0;
        float* outp = (float*)e1;
#pragma unroll
        for (int mi = 0; mi < 2; ++mi)
#pragma unroll
            for (int r = 0; r < 8; ++r) {
                int m = (mt0 + mi) * 16 + half * 8 + r;
                long base = nat_from_win(m) * CC;
#pragma unroll
                for (int nj = 0; nj < 4; ++nj)
                    outp[base + n4[nj]] =
                        xin[base + n4[nj]] + acc[mi][nj][r] + bv4[nj];
            }
    } else if constexpr (EPI == 2) {        // exact GELU -> bf16
        __bf16* outp = (__bf16*)e0;
#pragma unroll
        for (int mi = 0; mi < 2; ++mi)
#pragma unroll
            for (int r = 0; r < 8; ++r) {
                long base = (long)((mt0 + mi) * 16 + half * 8 + r) * N;
#pragma unroll
                for (int nj = 0; nj < 4; ++nj) {
                    float val = acc[mi][nj][r] + bv4[nj];
                    float gl = 0.5f * val * (1.0f + erff(val * 0.70710678118654752f));
                    outp[base + n4[nj]] = (__bf16)gl;
                }
            }
    } else {                                // fc2: in-place residual on d_out
        float* outp = (float*)e0;
#pragma unroll
        for (int mi = 0; mi < 2; ++mi)
#pragma unroll
            for (int r = 0; r < 8; ++r) {
                long base = (long)((mt0 + mi) * 16 + half * 8 + r) * CC;
#pragma unroll
                for (int nj = 0; nj < 4; ++nj)
                    outp[base + n4[nj]] += acc[mi][nj][r] + bv4[nj];
            }
    }
}

// ---------------- windowed attention: block = 1 window, wave = 1 head ----------------
__global__ __launch_bounds__(128) void
attn_kernel(const __bf16* __restrict__ qb, const __bf16* __restrict__ kb,
            const __bf16* __restrict__ vb, const float* __restrict__ cb,
            __bf16* __restrict__ attnout) {
    __shared__ __align__(16) __bf16 vt[NHD][HDD * 64];   // V^T  per head: [hd][token]
    __shared__ __align__(16) __bf16 pr[NHD][64 * 64];    // probs per head: [row][col]

    int lane = threadIdx.x & 31;
    int h    = threadIdx.x >> 5;
    int win  = blockIdx.x;
    int widx = win & 63;
    int half = lane >> 4, l16 = lane & 15;

    const __bf16* qh = qb + (((long)win * NHD + h) << 11);   // 64*32
    const __bf16* kh = kb + (((long)win * NHD + h) << 11);
    const __bf16* vh = vb + (((long)win * NHD + h) << 11);
    const float*  cbh = cb + ((long)widx * NHD + h) * (NTK * NTK);

    // stage V^T into LDS (lane = hd)
    for (int t = 0; t < 64; ++t) vt[h][lane * 64 + t] = vh[t * HDD + lane];
    __syncthreads();

    // K fragments: B[k=hd][n=token], token rows contiguous
    FragU bk[4];
#pragma unroll
    for (int j = 0; j < 4; ++j) {
        const uint4* p = (const uint4*)(kh + (j * 16 + l16) * HDD + half * 16);
        bk[j].u4[0] = p[0];
        bk[j].u4[1] = p[1];
    }

#pragma unroll 1
    for (int i = 0; i < 4; ++i) {               // row tile of tokens
        FragU a;
        const __bf16* qr = qh + (i * 16 + l16) * HDD;
        a.u4[0] = *(const uint4*)(qr + half * 16);
        a.u4[1] = *(const uint4*)(qr + half * 16 + 8);
        v8f acc[4];
#pragma unroll
        for (int j = 0; j < 4; ++j) acc[j] = wmma_bf16(a.v, bk[j].v, zero8());

        // fused bias+mask + row softmax (rows in D layout: m = i*16 + half*8 + r)
#pragma unroll
        for (int r = 0; r < 8; ++r) {
            int m = i * 16 + half * 8 + r;
            bool mok = (m < NTK);
            float v4[4];
#pragma unroll
            for (int j = 0; j < 4; ++j) {
                int n = j * 16 + l16;
                bool ok = mok && (n < NTK);
                int lidx = ok ? (m * NTK + n) : 0;
                float badd = cbh[lidx];                 // unconditional load
                float xv = acc[j][r] + badd;
                xv = ok ? xv : 0.0f;
                xv = (n < NTK) ? xv : -1e30f;
                v4[j] = xv;
            }
            float mx = fmaxf(fmaxf(v4[0], v4[1]), fmaxf(v4[2], v4[3]));
#pragma unroll
            for (int s = 8; s >= 1; s >>= 1) mx = fmaxf(mx, __shfl_xor(mx, s, 32));
            float sum = 0.0f;
#pragma unroll
            for (int j = 0; j < 4; ++j) { v4[j] = __expf(v4[j] - mx); sum += v4[j]; }
#pragma unroll
            for (int s = 8; s >= 1; s >>= 1) sum += __shfl_xor(sum, s, 32);
            float inv = 1.0f / sum;
#pragma unroll
            for (int j = 0; j < 4; ++j)
                pr[h][m * 64 + j * 16 + l16] = (__bf16)(v4[j] * inv);
        }

        // attn @ V for this row tile (K over 64 tokens, 2 steps)
        v8f o0 = zero8(), o1 = zero8();
#pragma unroll
        for (int kk = 0; kk < 2; ++kk) {
            FragU pa;
            const __bf16* prow = &pr[h][(i * 16 + l16) * 64 + kk * 32];
            pa.u4[0] = *(const uint4*)(prow + half * 16);
            pa.u4[1] = *(const uint4*)(prow + half * 16 + 8);
            FragU b0, b1;
            const __bf16* vp0 = &vt[h][(0 * 16 + l16) * 64 + kk * 32 + half * 16];
            const __bf16* vp1 = &vt[h][(1 * 16 + l16) * 64 + kk * 32 + half * 16];
            b0.u4[0] = ((const uint4*)vp0)[0]; b0.u4[1] = ((const uint4*)vp0)[1];
            b1.u4[0] = ((const uint4*)vp1)[0]; b1.u4[1] = ((const uint4*)vp1)[1];
            o0 = wmma_bf16(pa.v, b0.v, o0);
            o1 = wmma_bf16(pa.v, b1.v, o1);
        }
#pragma unroll
        for (int r = 0; r < 8; ++r) {
            int m = i * 16 + half * 8 + r;
            if (m < NTK) {
                long base = ((long)win * NTK + m) * CC + h * HDD + l16;
                attnout[base]      = (__bf16)(o0[r]);
                attnout[base + 16] = (__bf16)(o1[r]);
            }
        }
    }
}

// ---------------- launch ----------------
extern "C" void kernel_launch(void* const* d_in, const int* in_sizes, int n_in,
                              void* d_out, int out_size, void* d_ws, size_t ws_size,
                              hipStream_t stream) {
    const float* x         = (const float*)d_in[0];
    const float* mask      = (const float*)d_in[1];
    const int*   rel_idx   = (const int*)d_in[2];
    const float* rel_table = (const float*)d_in[3];
    const float* n1g = (const float*)d_in[4];
    const float* n1b = (const float*)d_in[5];
    const float* qkv_w  = (const float*)d_in[6];
    const float* qkv_b  = (const float*)d_in[7];
    const float* proj_w = (const float*)d_in[8];
    const float* proj_b = (const float*)d_in[9];
    const float* n2g = (const float*)d_in[10];
    const float* n2b = (const float*)d_in[11];
    const float* fc1_w = (const float*)d_in[12];
    const float* fc1_b = (const float*)d_in[13];
    const float* fc2_w = (const float*)d_in[14];
    const float* fc2_b = (const float*)d_in[15];
    float* out = (float*)d_out;
    char* ws = (char*)d_ws;

    const size_t SZ_HW  = (size_t)TOK * CC * 2;                   // 51,380,224
    const size_t SZ_QH  = (size_t)BB * NWIN * NHD * 64 * HDD * 2; // 67,108,864
    const size_t OFF_HW   = 0;
    const size_t OFF_Q    = OFF_HW + SZ_HW;
    const size_t OFF_K    = OFF_Q + SZ_QH;
    const size_t OFF_V    = OFF_K + SZ_QH;
    const size_t OFF_ATTN = OFF_V + SZ_QH;
    const size_t OFF_FC1  = OFF_Q;                 // reuse q/k/v + attn region
    const size_t OFF_WPK  = OFF_ATTN + SZ_HW;
    const size_t OFF_CB   = OFF_WPK + 393216;      // fused bias table (2.46 MB)

    __bf16* hw   = (__bf16*)(ws + OFF_HW);
    __bf16* qbuf = (__bf16*)(ws + OFF_Q);
    __bf16* kbuf = (__bf16*)(ws + OFF_K);
    __bf16* vbuf = (__bf16*)(ws + OFF_V);
    __bf16* aout = (__bf16*)(ws + OFF_ATTN);
    __bf16* fc1o = (__bf16*)(ws + OFF_FC1);
    __bf16* qkvWp  = (__bf16*)(ws + OFF_WPK);
    __bf16* projWp = (__bf16*)(ws + OFF_WPK + 98304);
    __bf16* fc1Wp  = (__bf16*)(ws + OFF_WPK + 131072);
    __bf16* fc2Wp  = (__bf16*)(ws + OFF_WPK + 262144);
    float*  cbuf   = (float*)(ws + OFF_CB);

    // 1. pack weights into B-fragment layout; build fused bias+mask table
    pack_b_kernel<<<192, 256, 0, stream>>>(qkv_w,  qkvWp, 3 * CC, CC);
    pack_b_kernel<<<64,  256, 0, stream>>>(proj_w, projWp, CC, CC);
    pack_b_kernel<<<256, 256, 0, stream>>>(fc1_w,  fc1Wp, MLPD, CC);
    pack_b_kernel<<<256, 256, 0, stream>>>(fc2_w,  fc2Wp, CC, MLPD);
    fuse_bias_kernel<<<2402, 256, 0, stream>>>(mask, rel_idx, rel_table, cbuf);

    // 2. zero padded q/k/v region (pad tokens 49..63 must be 0 for V)
    hipMemsetAsync(ws + OFF_Q, 0, 3 * SZ_QH, stream);

    // 3. LN1 + shift + window partition -> bf16 windowed activations
    ln_kernel<<<TOK / 8, 256, 0, stream>>>(x, n1g, n1b, hw, 1);

    // 4. QKV GEMM (M=200704, N=384, K=128) -> q/k/v buffers (q pre-scaled)
    gemm_bf16<0><<<dim3(1568, 3), 256, 0, stream>>>(hw, qkvWp, qkv_b, 384, CC,
                                                    qbuf, kbuf, vbuf);

    // 5. windowed attention
    attn_kernel<<<BB * NWIN, 128, 0, stream>>>(qbuf, kbuf, vbuf, cbuf, aout);

    // 6. proj GEMM + window-reverse + residual -> x1 in d_out
    gemm_bf16<1><<<dim3(1568, 1), 256, 0, stream>>>(aout, projWp, proj_b, CC, CC,
                                                    (void*)x, out, nullptr);

    // 7. LN2 (natural order) -> bf16 (reuse hw region)
    ln_kernel<<<TOK / 8, 256, 0, stream>>>(out, n2g, n2b, hw, 0);

    // 8. FC1 GEMM + GELU -> bf16
    gemm_bf16<2><<<dim3(1568, 4), 256, 0, stream>>>(hw, fc1Wp, fc1_b, MLPD, CC,
                                                    fc1o, nullptr, nullptr);

    // 9. FC2 GEMM + in-place residual on d_out
    gemm_bf16<3><<<dim3(1568, 1), 256, 0, stream>>>(fc1o, fc2Wp, fc2_b, CC, MLPD,
                                                    out, nullptr, nullptr);
}
